// Decoder_3908420239638
// MI455X (gfx1250) — matrix operand order repaired and checked by reading, not compiled, not run on hardware
//
#include <hip/hip_runtime.h>
#include <hip/hip_bf16.h>
#include <math.h>

#define BDIM 512
#define SDIM 16
#define HDIM 501
#define HP   512
#define H3   1503
#define N3   1536
#define NG   1024
#define CDIM 7
#define TCH  4      // t-steps batched per GEMM chunk

typedef __attribute__((ext_vector_type(16))) __bf16 v16bf;
typedef __attribute__((ext_vector_type(8)))  float  v8f;
typedef unsigned short u16;
typedef unsigned int   u32;

__device__ __forceinline__ u16 f2bf(float x) {
    u32 b = __builtin_bit_cast(u32, x);
    u32 r = b + 0x7FFFu + ((b >> 16) & 1u);
    return (u16)(r >> 16);
}
__device__ __forceinline__ float sigmoidf_(float x) { return 1.0f / (1.0f + __expf(-x)); }

// ---------------- small elementwise / setup kernels ----------------

__global__ void k_zero(float* p, int n) {
    int i = blockIdx.x * blockDim.x + threadIdx.x;
    if (i < n) p[i] = 0.f;
}

__global__ void k_g0(const float* b_gate, const float* b_map, float* g0) {
    int h = blockIdx.x * blockDim.x + threadIdx.x;
    if (h < HP) g0[h] = (h < HDIM) ? sigmoidf_(b_gate[h]) * b_map[h] : 0.f;
}

// rows x HDIM f32 -> rows x HP bf16, zero-padded K
__global__ void k_cvt_rows(const float* src, u16* dst, int rows) {
    int i = blockIdx.x * blockDim.x + threadIdx.x;
    if (i >= rows * HP) return;
    int b = i >> 9, h = i & 511;
    dst[i] = (h < HDIM) ? f2bf(src[b * HDIM + h]) : (u16)0;
}

// Pack weight matrix W (used as B = W^T, i.e. Bmat[k][n] = W[n][k]) into WMMA
// B-fragment order for v_wmma_f32_16x16x32_bf16:
//   dword d = ((kc*ntN + tileN)*32 + lane)*8 + v
//   n = tileN*16 + (lane&15), half = lane>>4, K(dword v) = kc*32 + 16*half + 2*v
// mode 0: W_hh (N=1536, Nreal=1503) ; mode 1: W_lin1 (N=512, Nreal=501)
// mode 2: [W_gate | W_map] concat along N (N=1024)
__global__ void k_pack(const float* s0, const float* s1, u16* dst, int N, int Nreal, int mode) {
    int ntN = N >> 4;
    int total = 16 * ntN * 32 * 8;
    int d = blockIdx.x * blockDim.x + threadIdx.x;
    if (d >= total) return;
    int v    = d & 7;
    int lane = (d >> 3) & 31;
    int tile = (d >> 8) % ntN;
    int kc   = (d >> 8) / ntN;
    int n    = tile * 16 + (lane & 15);
    int half = lane >> 4;
    int k0   = kc * 32 + 16 * half + 2 * v;
    float f0 = 0.f, f1 = 0.f;
    if (mode == 2) {
        const float* src = (n < 512) ? s0 : s1;
        int nn = (n < 512) ? n : (n - 512);
        if (nn < HDIM) {
            if (k0     < HDIM) f0 = src[nn * HDIM + k0];
            if (k0 + 1 < HDIM) f1 = src[nn * HDIM + k0 + 1];
        }
    } else {
        if (n < Nreal) {
            if (k0     < HDIM) f0 = s0[n * HDIM + k0];
            if (k0 + 1 < HDIM) f1 = s0[n * HDIM + k0 + 1];
        }
    }
    dst[2 * d]     = f2bf(f0);
    dst[2 * d + 1] = f2bf(f1);
}

// ------- WMMA GEMM: C(MxN) f32 = A(Mx512 bf16 row-major) * Bpk, 16x64 per wave -------
// Each wave owns 4 adjacent 16x16 N-tiles: one A fragment feeds 4 back-to-back WMMAs
// per 32-wide K chunk (K = 512 = 16 chunks).
__global__ void k_gemm4(const u16* __restrict__ A, const u16* __restrict__ Bpk,
                        float* __restrict__ C, int M, int N) {
    int ntN  = N >> 4;
    int ntN4 = N >> 6;
    int ntM  = M >> 4;
    int wave = blockIdx.x * (blockDim.x >> 5) + (threadIdx.x >> 5);
    if (wave >= ntM * ntN4) return;
    int tM = wave / ntN4, tN4 = wave % ntN4;
    int lane = threadIdx.x & 31;
    int half = lane >> 4;
    int m = tM * 16 + (lane & 15);
    const u32* A32 = (const u32*)A;
    const u32* B32 = (const u32*)Bpk;
    v8f acc0 = {}, acc1 = {}, acc2 = {}, acc3 = {};
    for (int kc = 0; kc < 16; ++kc) {
        union { v16bf v; u32 u[8]; } a, b0, b1, b2, b3;
        // A 16x32 layout (matches packed B): lanes0-15 K 0..15 in v0-7,
        // lanes16-31 K 16..31.
        int abase = (m << 9) + kc * 32 + 16 * half;
        #pragma unroll
        for (int q = 0; q < 4; ++q) {
            a.u[q]     = A32[(abase + 2 * q) >> 1];
            a.u[q + 4] = A32[(abase + 8 + 2 * q) >> 1];
        }
        const u32* bp = B32 + (((size_t)(kc * ntN + tN4 * 4) * 32 + lane) << 3);
        #pragma unroll
        for (int q = 0; q < 8; ++q) {
            b0.u[q] = bp[q];
            b1.u[q] = bp[256 + q];
            b2.u[q] = bp[512 + q];
            b3.u[q] = bp[768 + q];
        }
        acc0 = __builtin_amdgcn_wmma_f32_16x16x32_bf16(false, a.v, false, b0.v, (short)0, acc0, false, false);
        acc1 = __builtin_amdgcn_wmma_f32_16x16x32_bf16(false, a.v, false, b1.v, (short)0, acc1, false, false);
        acc2 = __builtin_amdgcn_wmma_f32_16x16x32_bf16(false, a.v, false, b2.v, (short)0, acc2, false, false);
        acc3 = __builtin_amdgcn_wmma_f32_16x16x32_bf16(false, a.v, false, b3.v, (short)0, acc3, false, false);
    }
    int nb0 = tN4 * 64 + (lane & 15);
    #pragma unroll
    for (int q = 0; q < 8; ++q) {
        size_t row = (size_t)(tM * 16 + q + 8 * half) * N;
        C[row + nb0]      = acc0[q];
        C[row + nb0 + 16] = acc1[q];
        C[row + nb0 + 32] = acc2[q];
        C[row + nb0 + 48] = acc3[q];
    }
}

// ---------------- per-step kernels ----------------

__global__ void k_gs0_ep(const float* gout, const float* b_lin1, float* gs0_f, u16* gs0_b) {
    int i = blockIdx.x * blockDim.x + threadIdx.x;
    if (i >= BDIM * HP) return;
    int b = i >> 9, h = i & 511;
    float v = (h < HDIM) ? gout[b * HP + h] + b_lin1[h] : 0.f;
    gs0_f[i] = v;
    gs0_b[i] = f2bf(v);
}

// gi[b][n] = one-hot(x_idx) @ W_ih.T + b_ih  (C=7 MACs)
__global__ void k_gi(const float* ne, const float* W_ih, const float* b_ih, float* gi, int idx) {
    int i = blockIdx.x * blockDim.x + threadIdx.x;
    if (i >= BDIM * H3) return;
    int b = i / H3, n = i % H3;
    const float* x = ne + (b * SDIM + idx) * CDIM;
    float a = b_ih[n];
    #pragma unroll
    for (int c = 0; c < CDIM; ++c) a += x[c] * W_ih[n * CDIM + c];
    gi[(size_t)b * N3 + n] = a;
}

// cumulative h_in(t) = 16*g0 + sum_{j>=idx-1-t} dep[b,idx,j]*(g_j - g0)
// computes `cnt` consecutive t values (t = t0..t0+cnt-1), rows tt*512+b
__global__ void k_hin4(const float* dep, const float* g0, const float* gmd,
                       float* accb, float* hin_f, u16* hin_b, int idx, int t0, int cnt) {
    int i = blockIdx.x * blockDim.x + threadIdx.x;
    if (i >= BDIM * HP) return;
    int b = i >> 9, h = i & 511;
    if (h < HDIM) {
        float a = (t0 == 0) ? 16.f * g0[h] : accb[i];
        for (int tt = 0; tt < cnt; ++tt) {
            int j = idx - 1 - (t0 + tt);
            a += dep[(b * SDIM + idx) * SDIM + j] * gmd[(((size_t)(j * BDIM + b)) << 9) + h];
            size_t o = (((size_t)(tt * BDIM + b)) << 9) + h;
            hin_f[o] = a;
            hin_b[o] = f2bf(a);
        }
        accb[i] = a;
    } else {
        for (int tt = 0; tt < cnt; ++tt) {
            size_t o = (((size_t)(tt * BDIM + b)) << 9) + h;
            hin_f[o] = 0.f;
            hin_b[o] = (u16)0;
        }
    }
}

// torch GRUCell combine over M rows (row r = tt*512 + b)
__global__ void k_gru(const float* gh, const float* gi, const float* b_hh,
                      const float* hin, float* hv, int M) {
    int i = blockIdx.x * blockDim.x + threadIdx.x;
    if (i >= M * HDIM) return;
    int r = i / HDIM, h = i % HDIM;
    const float* g = gh + (size_t)r * N3;
    const float* q = gi + (size_t)(r & 511) * N3;
    float ghr = g[h] + b_hh[h];
    float ghz = g[HDIM + h] + b_hh[HDIM + h];
    float ghn = g[2 * HDIM + h] + b_hh[2 * HDIM + h];
    float rr = sigmoidf_(q[h] + ghr);
    float u  = sigmoidf_(q[HDIM + h] + ghz);
    float nn = tanhf(q[2 * HDIM + h] + rr * ghn);
    hv[((size_t)r << 9) + h] = (1.f - u) * nn + u * hin[((size_t)r << 9) + h];
}

// e1p[t][b] = w1 . hv_t[b]  for chunk rows
__global__ void k_e1p(const float* hv, const float* W_edge, float* e1p, int t0, int cnt) {
    int i = blockIdx.x * blockDim.x + threadIdx.x;
    if (i >= cnt * BDIM) return;
    int tt = i >> 9, b = i & 511;
    const float* x = hv + (((size_t)(tt * BDIM + b)) << 9);
    float s = 0.f;
    for (int h = 0; h < HDIM; ++h) s += W_edge[h] * x[h];
    e1p[(t0 + tt) * BDIM + b] = s;
}

__global__ void k_fin(const float* hv, float* gs_cur, u16* hvfb) {
    int i = blockIdx.x * blockDim.x + threadIdx.x;
    if (i >= BDIM * HP) return;
    int h = i & 511;
    float v = (h < HDIM) ? hv[i] : 0.f;
    gs_cur[i] = v;
    hvfb[i]   = f2bf(v);
}

__global__ void k_dots(const float* hv, const float* W_edge, float* e1f, float* e2f, int idx) {
    int b = blockIdx.x * blockDim.x + threadIdx.x;
    if (b >= BDIM) return;
    const float* x = hv + ((size_t)b << 9);
    float s1 = 0.f, s2 = 0.f;
    for (int h = 0; h < HDIM; ++h) { s1 += W_edge[h] * x[h]; s2 += W_edge[HDIM + h] * x[h]; }
    e1f[idx * BDIM + b] = s1;
    e2f[idx * BDIM + b] = s2;
}

// edge[b, idx, vj] = (w1.hv_used + w2.hv_vj + b_edge >= 0)
__global__ void k_edge(const float* e1f, const float* e1p, const float* e2f,
                       const float* b_edge, float* dep_out, int idx) {
    int i = blockIdx.x * blockDim.x + threadIdx.x;
    if (i >= BDIM * idx) return;
    int b = i / idx, t = i % idx;
    int vj = idx - 1 - t;
    float used = (t == 0) ? e1f[(idx - 1) * BDIM + b] : e1p[(t - 1) * BDIM + b];
    float s = used + e2f[vj * BDIM + b] + b_edge[0];
    dep_out[(b * SDIM + idx) * SDIM + vj] = (s >= 0.f) ? 1.f : 0.f;
}

// gmd[idx][b][h] = sigmoid(gate)*(map) - g0
__global__ void k_gmep(const float* gmout, const float* b_gate, const float* b_map,
                       const float* g0, float* gmd, int idx) {
    int i = blockIdx.x * blockDim.x + threadIdx.x;
    if (i >= BDIM * HDIM) return;
    int b = i / HDIM, h = i % HDIM;
    float g = gmout[(size_t)b * NG + h] + b_gate[h];
    float m = gmout[(size_t)b * NG + 512 + h] + b_map[h];
    gmd[(((size_t)(idx * BDIM + b)) << 9) + h] = sigmoidf_(g) * m - g0[h];
}

__global__ void k_vert1(const float* gs, const float* W_vert, const float* b_vert, float* vtmp) {
    int i = blockIdx.x * blockDim.x + threadIdx.x;
    if (i >= BDIM * CDIM) return;
    int b = i / CDIM, c = i % CDIM;
    const float* x = gs + ((size_t)b << 9);
    float s = b_vert[c];
    for (int h = 0; h < HDIM; ++h) s += W_vert[c * HDIM + h] * x[h];
    vtmp[b * 8 + c] = s;
}

__global__ void k_vert2(const float* vtmp, float* enc_out, int idx) {
    int b = blockIdx.x * blockDim.x + threadIdx.x;
    if (b >= BDIM) return;
    const float* v = vtmp + b * 8;
    float mx = v[0];
    #pragma unroll
    for (int c = 1; c < CDIM; ++c) mx = fmaxf(mx, v[c]);
    float e[CDIM]; float s = 0.f;
    #pragma unroll
    for (int c = 0; c < CDIM; ++c) { e[c] = __expf(v[c] - mx); s += e[c]; }
    float inv = 1.f / s;
    #pragma unroll
    for (int c = 0; c < CDIM; ++c) enc_out[(b * SDIM + idx) * CDIM + c] = e[c] * inv;
}

// ---------------- host orchestration ----------------

extern "C" void kernel_launch(void* const* d_in, const int* in_sizes, int n_in,
                              void* d_out, int out_size, void* d_ws, size_t ws_size,
                              hipStream_t stream) {
    (void)in_sizes; (void)n_in; (void)out_size; (void)ws_size;
    const float* z      = (const float*)d_in[0];
    const float* dep    = (const float*)d_in[1];
    const float* ne     = (const float*)d_in[2];
    const float* W_lin1 = (const float*)d_in[3];
    const float* b_lin1 = (const float*)d_in[4];
    const float* W_vert = (const float*)d_in[5];
    const float* b_vert = (const float*)d_in[6];
    const float* W_edge = (const float*)d_in[7];
    const float* b_edge = (const float*)d_in[8];
    const float* W_gate = (const float*)d_in[9];
    const float* b_gate = (const float*)d_in[10];
    const float* W_map  = (const float*)d_in[11];
    const float* b_map  = (const float*)d_in[12];
    const float* W_ih   = (const float*)d_in[13];
    const float* b_ih   = (const float*)d_in[14];
    const float* W_hh   = (const float*)d_in[15];
    const float* b_hh   = (const float*)d_in[16];

    float* dep_out = (float*)d_out;                      // (B,S,S)
    float* enc_out = dep_out + BDIM * SDIM * SDIM;       // (B,S,C)

    char* base = (char*)d_ws;
    size_t off = 0;
    auto carve = [&](size_t bytes) -> void* {
        void* p = base + off;
        off += (bytes + 255) & ~(size_t)255;
        return p;
    };
    float* g0     = (float*)carve((size_t)HP * 4);
    float* gh     = (float*)carve((size_t)TCH * BDIM * N3 * 4);
    float* gi     = (float*)carve((size_t)BDIM * N3 * 4);
    float* gs0_f  = (float*)carve((size_t)BDIM * HP * 4);
    float* gs_cur = (float*)carve((size_t)BDIM * HP * 4);
    float* hv     = (float*)carve((size_t)TCH * BDIM * HP * 4);
    float* hin_f  = (float*)carve((size_t)TCH * BDIM * HP * 4);
    float* accb   = (float*)carve((size_t)BDIM * HP * 4);
    float* gmout  = (float*)carve((size_t)BDIM * NG * 4);
    float* gmd    = (float*)carve((size_t)SDIM * BDIM * HP * 4);
    float* e1f    = (float*)carve((size_t)SDIM * BDIM * 4);
    float* e2f    = (float*)carve((size_t)SDIM * BDIM * 4);
    float* e1p    = (float*)carve((size_t)SDIM * BDIM * 4);
    float* vtmp   = (float*)carve((size_t)BDIM * 8 * 4);
    u16* z_b      = (u16*)carve((size_t)BDIM * HP * 2);
    u16* gs0_b    = (u16*)carve((size_t)BDIM * HP * 2);
    u16* hin_b    = (u16*)carve((size_t)TCH * BDIM * HP * 2);
    u16* hvfb     = (u16*)carve((size_t)BDIM * HP * 2);
    u16* Wlin1pk  = (u16*)carve((size_t)16 * 32 * 512 * 2);
    u16* Whhpk    = (u16*)carve((size_t)16 * 96 * 512 * 2);
    u16* Wgmpk    = (u16*)carve((size_t)16 * 64 * 512 * 2);

    auto nb = [](int n) { return (n + 255) / 256; };
    // gemm4 launch: tiles = (M/16)*(N/64), 4 waves per 128-thread block
    auto gemm4 = [&](const u16* A, const u16* B, float* C, int M, int N) {
        int tiles = (M >> 4) * (N >> 6);
        k_gemm4<<<(tiles + 3) / 4, 128, 0, stream>>>(A, B, C, M, N);
    };

    int outN = BDIM * SDIM * SDIM + BDIM * SDIM * CDIM;
    k_zero<<<nb(outN), 256, 0, stream>>>(dep_out, outN);
    k_g0<<<nb(HP), 256, 0, stream>>>(b_gate, b_map, g0);
    k_cvt_rows<<<nb(BDIM * HP), 256, 0, stream>>>(z, z_b, BDIM);
    k_pack<<<nb(16 * 96 * 32 * 8), 256, 0, stream>>>(W_hh, nullptr, Whhpk, N3, H3, 0);
    k_pack<<<nb(16 * 32 * 32 * 8), 256, 0, stream>>>(W_lin1, nullptr, Wlin1pk, HP, HDIM, 1);
    k_pack<<<nb(16 * 64 * 32 * 8), 256, 0, stream>>>(W_gate, W_map, Wgmpk, NG, 0, 2);

    // graph_state0 = z @ W_lin1.T + b_lin1
    gemm4(z_b, Wlin1pk, gh, 512, 512);
    k_gs0_ep<<<nb(BDIM * HP), 256, 0, stream>>>(gh, b_lin1, gs0_f, gs0_b);

    for (int idx = 0; idx < SDIM; ++idx) {
        const float* gs = (idx == 0) ? gs0_f : gs_cur;
        k_vert1<<<nb(BDIM * CDIM), 256, 0, stream>>>(gs, W_vert, b_vert, vtmp);
        k_vert2<<<nb(BDIM), 256, 0, stream>>>(vtmp, enc_out, idx);
        k_gi<<<nb(BDIM * H3), 256, 0, stream>>>(ne, W_ih, b_ih, gi, idx);

        int ttL = 0;  // row offset of final t within its chunk
        if (idx == 0) {
            gemm4(gs0_b, Whhpk, gh, 512, N3);
            k_gru<<<nb(BDIM * HDIM), 256, 0, stream>>>(gh, gi, b_hh, gs0_f, hv, BDIM);
        } else {
            for (int t0 = 0; t0 < idx; t0 += TCH) {
                int c = (idx - t0 < TCH) ? (idx - t0) : TCH;
                k_hin4<<<nb(BDIM * HP), 256, 0, stream>>>(dep, g0, gmd, accb,
                                                          hin_f, hin_b, idx, t0, c);
                gemm4(hin_b, Whhpk, gh, c * BDIM, N3);
                k_gru<<<nb(c * BDIM * HDIM), 256, 0, stream>>>(gh, gi, b_hh, hin_f, hv,
                                                               c * BDIM);
                k_e1p<<<nb(c * BDIM), 256, 0, stream>>>(hv, W_edge, e1p, t0, c);
            }
            k_edge<<<nb(BDIM * idx), 256, 0, stream>>>(e1f, e1p, e2f, b_edge, dep_out, idx);
            ttL = (idx - 1) - TCH * ((idx - 1) / TCH);
        }

        const float* hv_last = hv + (((size_t)ttL * BDIM) << 9);
        k_fin<<<nb(BDIM * HP), 256, 0, stream>>>(hv_last, gs_cur, hvfb);
        k_dots<<<nb(BDIM), 256, 0, stream>>>(hv_last, W_edge, e1f, e2f, idx);
        gemm4(hvfb, Wgmpk, gmout, 512, NG);
        k_gmep<<<nb(BDIM * HDIM), 256, 0, stream>>>(gmout, b_gate, b_map, g0, gmd, idx);
    }
}
